// RefLocal_66640712565054
// MI455X (gfx1250) — matrix operand order repaired and checked by reading, not compiled, not run on hardware
//
#include <hip/hip_runtime.h>
#include <stdint.h>

// ---------------------------------------------------------------------------
// 5x5 windowed local attention on MI455X (gfx1250), wave32 + WMMA bf16.
//
// Roofline: ~170 MB HBM traffic @ 23.3 TB/s = ~7.3 us; ~1.06 GFLOP.
// Both GEMM stages run on v_wmma_f32_16x16x32_bf16 with a bf16 hi/lo
// truncation split (3 WMMAs per product, fp32 accumulate) for ~fp32 accuracy.
// This revision removes exec-mask branch ladders (clamped unconditional loads
// + zero-page pointer select), cheapens the bf16 split, and vectorizes the
// output stores by computing the value stage transposed.
// ---------------------------------------------------------------------------

typedef __attribute__((ext_vector_type(16))) __bf16 v16bf;
typedef __attribute__((ext_vector_type(8)))  float  v8f;

#define WMMA_BF16(a, b, c) \
  __builtin_amdgcn_wmma_f32_16x16x32_bf16(false, (a), false, (b), (short)0, (c), false, false)

static __device__ __forceinline__ __bf16 bf_from_top16(uint32_t u) {
  union { uint16_t s; __bf16 b; } o; o.s = (uint16_t)(u >> 16);
  return o.b;
}
// Truncation split: hi = f[31:16] as bf16, lo = bf16(f - hi). ~16 mantissa bits.
struct bfpair { __bf16 hi, lo; };
static __device__ __forceinline__ bfpair split_bf(float f) {
  union { float f; uint32_t u; } v; v.f = f;
  bfpair r;
  r.hi = bf_from_top16(v.u);
  union { uint32_t u; float f; } hf; hf.u = v.u & 0xFFFF0000u;
  union { float f; uint32_t u; } lo; lo.f = f - hf.f;
  r.lo = bf_from_top16(lo.u);
  return r;
}

static __device__ __forceinline__ v8f wmma3(v16bf aH, v16bf aL, v16bf bH, v16bf bL, v8f c) {
  // (aH+aL)*(bH+bL) ~= aH*bH + aL*bH + aH*bL   (drop aL*bL, below fp32 ulp)
  c = WMMA_BF16(aH, bH, c);
  c = WMMA_BF16(aL, bH, c);
  c = WMMA_BF16(aH, bL, c);
  return c;
}

namespace cfg {
constexpr int B = 8, H = 96, W = 96, C = 192, V = 96, P = 2; // 5x5 window, pad 2
constexpr int WT = W / 16;        // 6 query tiles per row
constexpr int WAVES = 4;          // waves per block
constexpr int TILES = B * H * WT; // 4608
}

__global__ __launch_bounds__(32 * cfg::WAVES, 1)
void local_attn_wmma(const float* __restrict__ mainp,
                     const float* __restrict__ refp,
                     const float* __restrict__ valp,
                     float* __restrict__ outp,
                     const float* __restrict__ zpage) {   // >=192 zero floats
  using namespace cfg;
  __shared__ float stage[WAVES][16 * 32];   // D-tile staging, 2 KB / wave

  const int lane  = threadIdx.x & 31;
  const int wv    = threadIdx.x >> 5;
  const int hiSel = lane >> 4;     // 0 for lanes 0-15, 1 for lanes 16-31
  const int l15   = lane & 15;

  const int tile = blockIdx.x * WAVES + wv;          // exact grid: no guard
  const int wt = tile % WT;
  const int h  = (tile / WT) % H;
  const int b  = tile / (WT * H);
  const int w0 = wt * 16;

  const float* refbase = refp + (size_t)b * H * W * C;
  const float* valbase = valp + (size_t)b * H * W * V;

  // Prefetch the 5 ref/val rows this tile touches (global_prefetch_b8).
  #pragma unroll
  for (int i = 0; i < 5; ++i) {
    int row = h + i - P;
    if ((unsigned)row < (unsigned)H) {
      int c0 = (w0 - P < 0) ? 0 : (w0 - P);
      __builtin_prefetch(refbase + ((size_t)row * W + c0) * C, 0, 1);
      __builtin_prefetch(valbase + ((size_t)row * W + c0) * V, 0, 1);
    }
  }

  // ---- Load query A-operand (16 queries x C), bf16 hi/lo, 6 K-chunks ----
  // A operand, this code's contiguous k order (same in B): lane L holds M=L&15; halves 0..7 -> K=c0..c0+7,
  // halves 8..15 -> K=c0+8..c0+15, c0 = kc*32 + 16*(L>>4).
  v16bf aH[6], aL[6];
  const float* mrow = mainp + ((size_t)(b * H + h) * W + (w0 + l15)) * C;
  #pragma unroll
  for (int kc = 0; kc < 6; ++kc) {
    const int c0 = kc * 32 + 16 * hiSel;
    #pragma unroll
    for (int t = 0; t < 8; ++t) {
      bfpair p0 = split_bf(mrow[c0 + t]);
      bfpair p1 = split_bf(mrow[c0 + 8 + t]);
      aH[kc][t]     = p0.hi;  aL[kc][t]     = p0.lo;
      aH[kc][t + 8] = p1.hi;  aL[kc][t + 8] = p1.lo;
    }
  }

  // ---- Logits: per row-offset i, D = Q(16xC) x K(Cx32) over key cols
  //      w0-2+n, n=0..31. OOB lanes read the zero page -> logit 0 (= padding).
  float lg[25];
  float* st = stage[wv];
  #pragma unroll
  for (int i = 0; i < 5; ++i) {
    const int row = h + i - P;
    const bool rowOK = (unsigned)row < (unsigned)H;

    // B operand, this code's contiguous k order (same as A): lane L holds N=L&15; halves t -> K = t + 16*(L>>4).
    const int col0 = w0 - P + l15;        // n = l15        (D0 columns)
    const int col1 = col0 + 16;           // n = 16 + l15   (D1 columns)
    const bool ok0 = rowOK && (unsigned)col0 < (unsigned)W;
    const bool ok1 = rowOK && (unsigned)col1 < (unsigned)W;
    // Per-lane pointer select: OOB lanes stream zeros from the zero page.
    const float* r0 = ok0 ? (refbase + ((size_t)row * W + col0) * C) : zpage;
    const float* r1 = ok1 ? (refbase + ((size_t)row * W + col1) * C) : zpage;

    v8f d0 = {}, d1 = {};
    #pragma unroll
    for (int kc = 0; kc < 6; ++kc) {
      const int c0 = kc * 32 + 16 * hiSel;
      const int cz = c0 & 127;   // keep zero-page reads inside 192 floats
      const int o0 = ok0 ? c0 : cz;
      const int o1 = ok1 ? c0 : cz;
      v16bf bH0, bL0, bH1, bL1;
      #pragma unroll
      for (int t = 0; t < 16; ++t) {
        bfpair p0 = split_bf(r0[o0 + t]);
        bfpair p1 = split_bf(r1[o1 + t]);
        bH0[t] = p0.hi;  bL0[t] = p0.lo;
        bH1[t] = p1.hi;  bL1[t] = p1.lo;
      }
      d0 = wmma3(aH[kc], aL[kc], bH0, bL0, d0);
      d1 = wmma3(aH[kc], aL[kc], bH1, bL1, d1);
    }

    // Stage D tiles to LDS: element (M,N) from lane (N | reg r), M = r + 8*(L>>4).
    const int mBase = 8 * hiSel;
    #pragma unroll
    for (int r = 0; r < 8; ++r) {
      st[(mBase + r) * 32 + l15]      = d0[r];
      st[(mBase + r) * 32 + 16 + l15] = d1[r];
    }
    __asm__ volatile("s_wait_dscnt 0" ::: "memory");
    // Lane's query q = l15 needs the diagonal band n = q..q+4.
    #pragma unroll
    for (int j = 0; j < 5; ++j)
      lg[i * 5 + j] = st[l15 * 32 + l15 + j];
    __asm__ volatile("s_wait_dscnt 0" ::: "memory");   // reads done before next i's stores
  }

  // ---- Softmax over the 25 logits (fp32, per lane; lanes 16-31 duplicate) ----
  float mx = lg[0];
  #pragma unroll
  for (int j = 1; j < 25; ++j) mx = fmaxf(mx, lg[j]);
  float at[25], s = 0.0f;
  #pragma unroll
  for (int j = 0; j < 25; ++j) { at[j] = __expf(lg[j] - mx); s += at[j]; }
  const float inv = 1.0f / s;
  #pragma unroll
  for (int j = 0; j < 25; ++j) at[j] *= inv;

  // ---- Value stage (transposed): outT(16v x 16q) += valT(16v x 32n) x bandT(32n x 16q)
  //      per row-offset i; lane's 8 accum regs = 8 consecutive v-channels.
  v8f o[6] = {};
  #pragma unroll
  for (int i = 0; i < 5; ++i) {
    const int row = h + i - P;
    const bool rowOK = (unsigned)row < (unsigned)H;
    const int rowC = rowOK ? row : 0;
    const float a0 = at[i * 5 + 0], a1 = at[i * 5 + 1], a2 = at[i * 5 + 2],
                a3 = at[i * 5 + 3], a4 = at[i * 5 + 4];

    // bandT (B layout): lane N = q = l15; halves t -> n = t + 16*(L>>4);
    // B[n][q] = attn[n-q] for n-q in [0,4], else 0.
    v16bf bandH, bandL;
    #pragma unroll
    for (int t = 0; t < 16; ++t) {
      const int n = t + 16 * hiSel;
      const int d = n - l15;
      float v = (d == 0) ? a0 : (d == 1) ? a1 : (d == 2) ? a2
              : (d == 3) ? a3 : (d == 4) ? a4 : 0.0f;
      bfpair p = split_bf(v);
      bandH[t] = p.hi;  bandL[t] = p.lo;
    }

    const float* vrow = valbase + (size_t)rowC * W * V;
    #pragma unroll
    for (int vc = 0; vc < 6; ++vc) {
      // valT (A layout): lane M = v = vc*16 + l15; halves t -> n = t + 16*(L>>4).
      const int v = vc * 16 + l15;
      v16bf vH, vL;
      #pragma unroll
      for (int t = 0; t < 16; ++t) {
        const int n = t + 16 * hiSel;
        const int col = w0 - P + n;
        const bool ok = rowOK && (unsigned)col < (unsigned)W;
        const int colC = ((unsigned)col < (unsigned)W) ? col : 0;
        float f = vrow[(size_t)colC * V + v];      // clamped, unconditional load
        f = ok ? f : 0.0f;                         // one v_cndmask, no branch
        bfpair p = split_bf(f);
        vH[t] = p.hi;  vL[t] = p.lo;
      }
      o[vc] = wmma3(vH, vL, bandH, bandL, o[vc]);
    }
  }

  // ---- Store outT: element (M=v', N=q) -> out[b,h,w0+q, vc*16 + 8*hiSel + r],
  //      8 consecutive channels per lane => two float4 stores per vc.
  float* orow = outp + (((size_t)(b * H + h) * W) + w0 + l15) * V;
  #pragma unroll
  for (int vc = 0; vc < 6; ++vc) {
    const int v0 = vc * 16 + 8 * hiSel;
    float4 lo4 = make_float4(o[vc][0], o[vc][1], o[vc][2], o[vc][3]);
    float4 hi4 = make_float4(o[vc][4], o[vc][5], o[vc][6], o[vc][7]);
    *reinterpret_cast<float4*>(orow + v0)     = lo4;
    *reinterpret_cast<float4*>(orow + v0 + 4) = hi4;
  }
}

extern "C" void kernel_launch(void* const* d_in, const int* in_sizes, int n_in,
                              void* d_out, int out_size, void* d_ws, size_t ws_size,
                              hipStream_t stream) {
  (void)in_sizes; (void)n_in; (void)out_size;
  const float* mn = (const float*)d_in[0];   // main      [8,96,96,192]
  const float* rf = (const float*)d_in[1];   // ref       [8,96,96,192]
  const float* vl = (const float*)d_in[2];   // ref_value [8,96,96,96]
  float* out = (float*)d_out;                // [8,96,96,96]

  // Zero page for OOB key reads (192 floats needed; zero 4 KB).
  size_t zbytes = ws_size < 4096 ? ws_size : (size_t)4096;
  hipMemsetAsync(d_ws, 0, zbytes, stream);
  const float* zpage = (const float*)d_ws;

  const int blocks = cfg::TILES / cfg::WAVES;  // 4608 / 4 = 1152, exact
  local_attn_wmma<<<blocks, 32 * cfg::WAVES, 0, stream>>>(mn, rf, vl, out, zpage);
}
